// MEAM_44787918963464
// MI455X (gfx1250) — matrix operand order repaired and checked by reading, not compiled, not run on hardware
//
#include <hip/hip_runtime.h>
#include <cstdint>
#include <cstddef>

typedef _Float16 h16;
typedef __attribute__((ext_vector_type(16))) _Float16 v16h;
typedef __attribute__((ext_vector_type(8)))  _Float16 v8h;
typedef __attribute__((ext_vector_type(8)))  float    v8f;

#define Bn   32
#define Cc   64
#define Nn   307
#define Tt   24
#define NT   (Nn*Tt)      /* 7368 */
#define NTPB 7424         /* 7368 padded to 64 (4 tiles of 16) for f16 B operands */
#define COUT 192
#define NP   320          /* N padded to 16 for graph attention */
#define EG   (64*Tt)      /* 1536 = (o,t) fused dim for g branch */

// ---------------------------------------------------------------------------
// LayerNorm over T; also emit f16 transposed copies hT[b][j][c], xT[b][j][c]
// ---------------------------------------------------------------------------
__global__ void k_ln(const float* __restrict__ x, const float* __restrict__ g,
                     const float* __restrict__ bb, float* __restrict__ h,
                     h16* __restrict__ hT, h16* __restrict__ xT)
{
    int tid = blockIdx.x * blockDim.x + threadIdx.x;
    if (tid >= Bn * Cc * Nn) return;
    int n = tid % Nn;
    int c = (tid / Nn) % Cc;
    int b = tid / (Cc * Nn);
    const float* xr = x + (size_t)tid * Tt;
    float v[Tt], s = 0.f, ss = 0.f;
#pragma unroll
    for (int t = 0; t < Tt; ++t) { v[t] = xr[t]; s += v[t]; ss += v[t] * v[t]; }
    float mu = s * (1.0f / Tt);
    float var = ss * (1.0f / Tt) - mu * mu;
    float rs = rsqrtf(var + 1e-5f);
    float* hr = h + (size_t)tid * Tt;
    size_t jb = (size_t)b * NTPB + (size_t)n * Tt;
#pragma unroll
    for (int t = 0; t < Tt; ++t) {
        float y = (v[t] - mu) * rs * g[t] + bb[t];
        hr[t] = y;
        hT[(jb + t) * 64 + c] = (h16)y;
        xT[(jb + t) * 64 + c] = (h16)v[t];
    }
}

// ---------------------------------------------------------------------------
// f16 weight prep: res_W, g_W, conv weights concatenated [cur | prev tap],
// fused per-channel bias (res_b + cc_b / tc_b1)
// ---------------------------------------------------------------------------
__global__ void k_prep(const float* __restrict__ res_W, const float* __restrict__ res_b,
                       const float* __restrict__ cc_b,  const float* __restrict__ tc_W0,
                       const float* __restrict__ tc_W1, const float* __restrict__ tc_b1,
                       const float* __restrict__ g_W,
                       h16* __restrict__ resW16, float* __restrict__ resbias,
                       h16* __restrict__ gW16, h16* __restrict__ W0cat, h16* __restrict__ W1cat)
{
    int tid = threadIdx.x;
    for (int i = tid; i < 192 * 64; i += 256) resW16[i] = (h16)res_W[i];
    for (int i = tid; i < 192; i += 256)
        resbias[i] = res_b[i] + (i < 64 ? cc_b[i] : (i < 128 ? tc_b1[i - 64] : 0.f));
    for (int i = tid; i < 64 * 64; i += 256) {
        gW16[i] = (h16)g_W[i];
        int o = i >> 6, c = i & 63;
        W0cat[o * 128 + c]      = (h16)tc_W0[i * 2 + 1];   // current tap
        W0cat[o * 128 + 64 + c] = (h16)tc_W0[i * 2 + 0];   // t-d tap
        W1cat[o * 128 + c]      = (h16)tc_W1[i * 2 + 1];
        W1cat[o * 128 + 64 + c] = (h16)tc_W1[i * 2 + 0];
    }
}

// ---------------------------------------------------------------------------
// Stage 1: per (b,c,n): reduce over t with ca_W1 / ga_W1
// ---------------------------------------------------------------------------
__global__ void k_s1(const float* __restrict__ h, const float* __restrict__ caW1,
                     const float* __restrict__ gaW1,
                     float* __restrict__ bcn_c, float* __restrict__ bcn_g)
{
    int tid = blockIdx.x * blockDim.x + threadIdx.x;
    if (tid >= Bn * Cc * Nn) return;
    const float* hr = h + (size_t)tid * Tt;
    float a = 0.f, g = 0.f;
#pragma unroll
    for (int t = 0; t < Tt; ++t) { float hv = hr[t]; a += hv * caW1[t]; g += hv * gaW1[t]; }
    bcn_c[tid] = a; bcn_g[tid] = g;
}

// ---------------------------------------------------------------------------
// Stage 2: per (b,c,t): lhs_c (bcn_c x ca_W2), rhs_c (W3 x h over n),
//                       bct_t (ta_W1 x h over n)
// ---------------------------------------------------------------------------
__global__ void k_s2(const float* __restrict__ h, const float* __restrict__ bcn_c,
                     const float* __restrict__ caW2, const float* __restrict__ caW3,
                     const float* __restrict__ taW1,
                     float* __restrict__ lhs_c, float* __restrict__ rhs_c,
                     float* __restrict__ bct_t)
{
    int tid = blockIdx.x * blockDim.x + threadIdx.x;
    if (tid >= Bn * Cc * Tt) return;
    int t = tid % Tt;
    int c = (tid / Tt) % Cc;
    int b = tid / (Tt * Cc);
    const float* bc = bcn_c + ((size_t)b * Cc + c) * Nn;
    const float* hb = h + (((size_t)b * Cc + c) * Nn) * Tt + t;
    float aL = 0.f, aR = 0.f, aT = 0.f;
    for (int n = 0; n < Nn; ++n) {
        float hv = hb[(size_t)n * Tt];
        aL += bc[n] * caW2[n * Tt + t];
        aR += caW3[n] * hv;
        aT += taW1[n] * hv;
    }
    lhs_c[tid] = aL; rhs_c[tid] = aR; bct_t[tid] = aT;
}

// ---------------------------------------------------------------------------
// Stage 3: per (b,n,t): lhs_t, rhs_t, lhs_g, rhs_g  (all 64-long c loops)
// ---------------------------------------------------------------------------
__global__ void k_s3(const float* __restrict__ h, const float* __restrict__ bcn_g,
                     const float* __restrict__ bct_t, const float* __restrict__ taW2,
                     const float* __restrict__ taW3, const float* __restrict__ gaW2,
                     const float* __restrict__ gaW3,
                     float* __restrict__ lhs_t, float* __restrict__ rhs_t,
                     float* __restrict__ lhs_g, float* __restrict__ rhs_g)
{
    int tid = blockIdx.x * blockDim.x + threadIdx.x;
    if (tid >= Bn * Nn * Tt) return;
    int t = tid % Tt;
    int n = (tid / Tt) % Nn;
    int b = tid / (Tt * Nn);
    float aLT = 0.f, aRT = 0.f, aLG = 0.f, aRG = 0.f;
    for (int c = 0; c < Cc; ++c) {
        float hv = h[((((size_t)b * Cc + c) * Nn) + n) * Tt + t];
        aRT += taW3[c] * hv;
        aRG += gaW3[c] * hv;
        aLT += bct_t[((size_t)b * Cc + c) * Tt + t] * taW2[c * Nn + n];
        aLG += bcn_g[((size_t)b * Cc + c) * Nn + n] * gaW2[c * Tt + t];
    }
    rhs_t[tid] = aRT;                           // [B,N,T]
    rhs_g[tid] = aRG;                           // [B,N,T]
    lhs_g[tid] = aLG;                           // [B,N,T]
    lhs_t[((size_t)b * Tt + t) * Nn + n] = aLT; // [B,T,N]
}

// ---------------------------------------------------------------------------
// Channel attention scores + softmax over d (64 wide); block = (b,c)
// ---------------------------------------------------------------------------
__global__ void k_att_c(const float* __restrict__ lhs, const float* __restrict__ rhs,
                        float* __restrict__ att)
{
    int b = blockIdx.x / Cc, c = blockIdx.x % Cc, d = threadIdx.x;
    const float* lr = lhs + ((size_t)b * Cc + c) * Tt;
    const float* rr = rhs + ((size_t)b * Cc + d) * Tt;
    float s = 0.f;
#pragma unroll
    for (int t = 0; t < Tt; ++t) s += lr[t] * rr[t];
    __shared__ float red[64];
    __shared__ float s_mx, s_sm;
    red[d] = s; __syncthreads();
    if (d == 0) { float m = -3.0e38f; for (int i = 0; i < 64; ++i) m = fmaxf(m, red[i]); s_mx = m; }
    __syncthreads();
    float p = expf(s - s_mx);
    red[d] = p; __syncthreads();
    if (d == 0) { float sm = 0.f; for (int i = 0; i < 64; ++i) sm += red[i]; s_sm = sm; }
    __syncthreads();
    att[((size_t)b * Cc + c) * Cc + d] = p / s_sm;
}

// Mc[b] = cc_W @ att_c[b]  (64x64, f16 output for WMMA A operand)
__global__ void k_Mc(const float* __restrict__ cc_W, const float* __restrict__ att,
                     h16* __restrict__ Mc16)
{
    int tid = blockIdx.x * blockDim.x + threadIdx.x;
    if (tid >= Bn * 64 * 64) return;
    int c = tid % 64, o = (tid / 64) % 64, b = tid / 4096;
    float a = 0.f;
    for (int d = 0; d < 64; ++d) a += cc_W[o * 64 + d] * att[((size_t)b * 64 + d) * 64 + c];
    Mc16[(size_t)b * 4096 + o * 64 + c] = (h16)a;
}

// ---------------------------------------------------------------------------
// Temporal attention scores + softmax (24x24); block = (b,t)
// ---------------------------------------------------------------------------
__global__ void k_att_t(const float* __restrict__ lhs, const float* __restrict__ rhs,
                        float* __restrict__ att)
{
    int b = blockIdx.x / Tt, t = blockIdx.x % Tt, s = threadIdx.x;
    float sc = -3.0e38f;
    if (s < Tt) {
        float a = 0.f;
        const float* lr = lhs + ((size_t)b * Tt + t) * Nn;
        for (int n = 0; n < Nn; ++n) a += lr[n] * rhs[((size_t)b * Nn + n) * Tt + s];
        sc = a;
    }
    __shared__ float red[32];
    __shared__ float s_mx, s_sm;
    red[s] = sc; __syncthreads();
    if (s == 0) { float m = -3.0e38f; for (int i = 0; i < Tt; ++i) m = fmaxf(m, red[i]); s_mx = m; }
    __syncthreads();
    float p = expf(sc - s_mx);
    red[s] = p; __syncthreads();
    if (s == 0) { float sm = 0.f; for (int i = 0; i < Tt; ++i) sm += red[i]; s_sm = sm; }
    __syncthreads();
    if (s < Tt) att[((size_t)b * Tt + t) * Tt + s] = p / s_sm;
}

// ---------------------------------------------------------------------------
// Graph attention: scores, adjacency mask, softmax over m; f16 padded output
// block = (b,n), 320 threads
// ---------------------------------------------------------------------------
__global__ void k_att_g(const float* __restrict__ lhs, const float* __restrict__ rhs,
                        const int* __restrict__ adj, h16* __restrict__ att16)
{
    int b = blockIdx.x / Nn, n = blockIdx.x % Nn, m = threadIdx.x;
    float sc = -3.0e38f;
    if (m < Nn) {
        float a = 0.f;
        const float* lr = lhs + ((size_t)b * Nn + n) * Tt;
        const float* rr = rhs + ((size_t)b * Nn + m) * Tt;
#pragma unroll
        for (int t = 0; t < Tt; ++t) a += lr[t] * rr[t];
        sc = (adj[n * Nn + m] > 0) ? a : -1.0e30f;
    }
    __shared__ float red[NP];
    __shared__ float s_mx, s_sm;
    red[m] = sc; __syncthreads();
    if (m == 0) { float mx = -3.0e38f; for (int i = 0; i < NP; ++i) mx = fmaxf(mx, red[i]); s_mx = mx; }
    __syncthreads();
    float p = expf(sc - s_mx);   // exactly 0 for padded lanes
    red[m] = p; __syncthreads();
    if (m == 0) { float sm = 0.f; for (int i = 0; i < NP; ++i) sm += red[i]; s_sm = sm; }
    __syncthreads();
    att16[((size_t)b * NP + n) * NP + m] = (h16)(p / s_sm);
}

// ---------------------------------------------------------------------------
// Temporal attention apply: tout[c,n,t] = sum_s att[t,s]*h[c,n,s].
// Writes f16 into tcatT[b][j][0:64] and the d=1 shifted copy into [64:128].
// ---------------------------------------------------------------------------
__global__ void k_tapply(const float* __restrict__ h, const float* __restrict__ att_t,
                         h16* __restrict__ tcatT)
{
    const int ROWS = Cc * Nn;               // 19648
    const int BLK = (ROWS + 255) / 256;     // 77
    int b = blockIdx.x / BLK;
    int row = (blockIdx.x % BLK) * 256 + threadIdx.x;
    __shared__ float att[Tt * Tt];
    for (int i = threadIdx.x; i < Tt * Tt; i += 256) att[i] = att_t[(size_t)b * Tt * Tt + i];
    __syncthreads();
    if (row >= ROWS) return;
    int c = row / Nn, n = row % Nn;
    const float* hr = h + ((size_t)b * ROWS + row) * Tt;
    float xr[Tt];
#pragma unroll
    for (int t = 0; t < Tt; ++t) xr[t] = hr[t];
    h16* base = tcatT + (size_t)b * NTPB * 128;
#pragma unroll
    for (int t = 0; t < Tt; ++t) {
        float a = 0.f;
#pragma unroll
        for (int s = 0; s < Tt; ++s) a += att[t * Tt + s] * xr[s];
        size_t j = (size_t)n * Tt + t;
        base[j * 128 + c] = (h16)a;
        if (t < Tt - 1) base[(j + 1) * 128 + 64 + c] = (h16)a;  // t-1 tap for j+1
    }
}

// ---------------------------------------------------------------------------
// Generic f16 WMMA GEMM: wave computes a 16x16 tile; 4 col tiles per wave.
// Column dim is padded to a multiple of 64 so the k/q main loop is branch-free
// and every WMMA executes with EXEC all-ones (ISA 7.12 requirement).
// Wave/tile coordinates go through readfirstlane so control flow is scalar.
// A: [M][K] row-major f16.  Bt: [Ncol][K] row-major f16 (i.e. B transposed).
// MODE 0: c-branch  (A per-batch Mc16)      -> f32 d_out rows [0,64)
// MODE 1: g_W @ h                           -> f16 scatter HgB[e][m]
// MODE 2: conv0 (+tc_b0)                    -> f16 y0catT + d=2 shifted copy
// MODE 3: conv1                             -> f32 d_out rows [64,128)
// MODE 4: residual: +bias +d_out, ReLU      -> f32 d_out (all rows)
// MODE 5: att_g @ HgB (A per-batch, 320^2)  -> f32 scatter d_out rows [128,192)
// ---------------------------------------------------------------------------
template <int MODE>
__global__ __launch_bounds__(256) void gemm_k(const h16* __restrict__ A,
                                              const h16* __restrict__ Bt,
                                              const float* __restrict__ bias,
                                              float* __restrict__ dout,
                                              h16* __restrict__ outH)
{
    const int K     = (MODE == 2 || MODE == 3) ? 128 : (MODE == 5 ? 320 : 64);
    const int M     = (MODE == 4) ? 192 : (MODE == 5 ? 320 : 64);
    const int Mt    = M / 16;
    const int NCOLP = (MODE == 5) ? EG : NTPB;   // padded: multiple of 64
    const int NG    = NCOLP / 64;                // full 4-tile groups only
    const size_t Astride = (MODE == 0) ? (size_t)64 * 64 : (MODE == 5 ? (size_t)NP * NP : 0);
    const size_t Bstride = (MODE == 5) ? (size_t)EG * NP : (size_t)NTPB * K;

    int lane = threadIdx.x & 31;
    // wave id as an SGPR -> scalar control flow, no EXEC masking around WMMA
    int wid = __builtin_amdgcn_readfirstlane((blockIdx.x * blockDim.x + threadIdx.x) >> 5);
    int tasksPerB = Mt * NG;
    int b = wid / tasksPerB;
    if (b >= Bn) return;
    int r  = wid % tasksPerB;
    int mt = r / NG, ng = r % NG;
    int row0 = mt * 16;
    int col0base = ng * 64;

    const h16* Ab = A + (size_t)b * Astride;
    const h16* Bb = Bt + (size_t)b * Bstride;

    v8f acc[4] = {};
    int am   = row0 + (lane & 15);
    int aoff = (lane < 16) ? 0 : 16;   // A: lanes 0-15 hold K 0..15
    int boff = (lane < 16) ? 0 : 16;   // B: lanes 0-15 hold K 0..15

    for (int k0 = 0; k0 < K; k0 += 32) {
        const h16* ap = Ab + (size_t)am * K + k0 + aoff;
        v8h alo = *(const v8h*)ap;
        v8h ahi = *(const v8h*)(ap + 8);
        v16h af;
#pragma unroll
        for (int i = 0; i < 8; ++i) { af[i] = alo[i]; af[8 + i] = ahi[i]; }
        const h16* bbase = Bb + (size_t)(col0base + (lane & 15)) * K + k0 + boff;
#pragma unroll
        for (int q = 0; q < 4; ++q) {
            v16h bf = *(const v16h*)(bbase + (size_t)q * 16 * K);
            acc[q] = __builtin_amdgcn_wmma_f32_16x16x32_f16(
                false, af, false, bf, (short)0, acc[q], false, false);
        }
    }

    int cf = lane & 15;
    int rb = (lane >> 4) * 8;   // C/D layout: rows 0-7 lanes 0-15, rows 8-15 lanes 16-31
#pragma unroll
    for (int q = 0; q < 4; ++q) {
        int j = col0base + q * 16 + cf;
#pragma unroll
        for (int rr = 0; rr < 8; ++rr) {
            int m = row0 + rb + rr;
            float v = acc[q][rr];
            if (MODE == 0) {
                if (j < NT) dout[((size_t)b * COUT + m) * NT + j] = v;
            } else if (MODE == 1) {
                if (j < NT) {
                    int mm = j / Tt, tt = j - mm * Tt;
                    outH[(size_t)b * EG * NP + (size_t)(m * Tt + tt) * NP + mm] = (h16)v;
                }
            } else if (MODE == 2) {
                if (j < NT) {
                    float y = v + bias[m];
                    h16* yb = outH + (size_t)b * NTPB * 128;
                    yb[(size_t)j * 128 + m] = (h16)y;
                    int tt = j % Tt;
                    if (tt < Tt - 2) yb[(size_t)(j + 2) * 128 + 64 + m] = (h16)y; // t-2 tap
                }
            } else if (MODE == 3) {
                if (j < NT) dout[((size_t)b * COUT + 64 + m) * NT + j] = v;
            } else if (MODE == 5) {
                if (m < Nn) {
                    int o3 = j / Tt, tt = j - o3 * Tt;
                    dout[(((size_t)b * COUT + 128 + o3) * Nn + m) * Tt + tt] = v;
                }
            } else { // MODE 4
                if (j < NT) {
                    size_t idx = ((size_t)b * COUT + m) * NT + j;
                    float y = v + bias[m] + dout[idx];
                    dout[idx] = fmaxf(y, 0.f);
                }
            }
        }
    }
}

// ---------------------------------------------------------------------------
extern "C" void kernel_launch(void* const* d_in, const int* in_sizes, int n_in,
                              void* d_out, int out_size, void* d_ws, size_t ws_size,
                              hipStream_t stream)
{
    const float* x     = (const float*)d_in[0];
    const int*   adj   = (const int*)  d_in[1];
    const float* ln_g  = (const float*)d_in[2];
    const float* ln_b  = (const float*)d_in[3];
    const float* res_W = (const float*)d_in[4];
    const float* res_b = (const float*)d_in[5];
    const float* caW1  = (const float*)d_in[6];
    const float* caW2  = (const float*)d_in[7];
    const float* caW3  = (const float*)d_in[8];
    const float* cc_W  = (const float*)d_in[9];
    const float* cc_b  = (const float*)d_in[10];
    const float* taW1  = (const float*)d_in[11];
    const float* taW2  = (const float*)d_in[12];
    const float* taW3  = (const float*)d_in[13];
    const float* tc_W0 = (const float*)d_in[14];
    const float* tc_b0 = (const float*)d_in[15];
    const float* tc_W1 = (const float*)d_in[16];
    const float* tc_b1 = (const float*)d_in[17];
    const float* gaW1  = (const float*)d_in[18];
    const float* gaW2  = (const float*)d_in[19];
    const float* gaW3  = (const float*)d_in[20];
    const float* g_W   = (const float*)d_in[21];
    float* dout = (float*)d_out;

    // ---- workspace carve-up (256B aligned) ----
    char* p = (char*)d_ws;
    auto alloc = [&](size_t bytes) -> void* {
        void* r = (void*)p;
        p += (bytes + 255) & ~(size_t)255;
        return r;
    };
    float* h      = (float*)alloc((size_t)Bn * Cc * Nn * Tt * 4);
    h16*  hT16    = (h16*)  alloc((size_t)Bn * NTPB * 64 * 2);
    h16*  xT16    = (h16*)  alloc((size_t)Bn * NTPB * 64 * 2);
    float* bcn_c  = (float*)alloc((size_t)Bn * Cc * Nn * 4);
    float* bcn_g  = (float*)alloc((size_t)Bn * Cc * Nn * 4);
    float* lhs_c  = (float*)alloc((size_t)Bn * Cc * Tt * 4);
    float* rhs_c  = (float*)alloc((size_t)Bn * Cc * Tt * 4);
    float* bct_t  = (float*)alloc((size_t)Bn * Cc * Tt * 4);
    float* lhs_t  = (float*)alloc((size_t)Bn * Tt * Nn * 4);
    float* rhs_t  = (float*)alloc((size_t)Bn * Nn * Tt * 4);
    float* lhs_g  = (float*)alloc((size_t)Bn * Nn * Tt * 4);
    float* rhs_g  = (float*)alloc((size_t)Bn * Nn * Tt * 4);
    float* att_c  = (float*)alloc((size_t)Bn * 64 * 64 * 4);
    float* att_t  = (float*)alloc((size_t)Bn * Tt * Tt * 4);
    h16*  attg16  = (h16*)  alloc((size_t)Bn * NP * NP * 2);
    h16*  Mc16    = (h16*)  alloc((size_t)Bn * 64 * 64 * 2);
    h16*  HgB     = (h16*)  alloc((size_t)Bn * EG * NP * 2);
    h16*  tcatT   = (h16*)  alloc((size_t)Bn * NTPB * 128 * 2);
    h16*  y0catT  = (h16*)  alloc((size_t)Bn * NTPB * 128 * 2);
    h16*  resW16  = (h16*)  alloc((size_t)192 * 64 * 2);
    float* resbias= (float*)alloc((size_t)192 * 4);
    h16*  gW16    = (h16*)  alloc((size_t)64 * 64 * 2);
    h16*  W0cat16 = (h16*)  alloc((size_t)64 * 128 * 2);
    h16*  W1cat16 = (h16*)  alloc((size_t)64 * 128 * 2);

    // zero-fill padded f16 operands (edge tiles / shift slots multiply by 0)
    hipMemsetAsync(hT16,   0, (size_t)Bn * NTPB * 64 * 2, stream);
    hipMemsetAsync(xT16,   0, (size_t)Bn * NTPB * 64 * 2, stream);
    hipMemsetAsync(attg16, 0, (size_t)Bn * NP * NP * 2, stream);
    hipMemsetAsync(HgB,    0, (size_t)Bn * EG * NP * 2, stream);
    hipMemsetAsync(tcatT,  0, (size_t)Bn * NTPB * 128 * 2, stream);
    hipMemsetAsync(y0catT, 0, (size_t)Bn * NTPB * 128 * 2, stream);

    k_prep<<<1, 256, 0, stream>>>(res_W, res_b, cc_b, tc_W0, tc_W1, tc_b1, g_W,
                                  resW16, resbias, gW16, W0cat16, W1cat16);

    const int RBCN = Bn * Cc * Nn; // 628736
    k_ln<<<(RBCN + 255) / 256, 256, 0, stream>>>(x, ln_g, ln_b, h, hT16, xT16);
    k_s1<<<(RBCN + 255) / 256, 256, 0, stream>>>(h, caW1, gaW1, bcn_c, bcn_g);
    k_s2<<<(Bn * Cc * Tt + 255) / 256, 256, 0, stream>>>(h, bcn_c, caW2, caW3, taW1,
                                                         lhs_c, rhs_c, bct_t);
    k_s3<<<(Bn * Nn * Tt + 255) / 256, 256, 0, stream>>>(h, bcn_g, bct_t, taW2, taW3,
                                                         gaW2, gaW3,
                                                         lhs_t, rhs_t, lhs_g, rhs_g);
    k_att_c<<<Bn * Cc, 64, 0, stream>>>(lhs_c, rhs_c, att_c);
    k_att_t<<<Bn * Tt, 32, 0, stream>>>(lhs_t, rhs_t, att_t);
    k_att_g<<<Bn * Nn, NP, 0, stream>>>(lhs_g, rhs_g, adj, attg16);
    k_Mc<<<(Bn * 64 * 64 + 255) / 256, 256, 0, stream>>>(cc_W, att_c, Mc16);

    auto blocks8 = [](long waves) -> unsigned { return (unsigned)((waves + 7) / 8); };
    const long W64 = (long)Bn * 4 * (NTPB / 64);   // M=64 over padded cols (116 groups)
    // c-branch: (cc_W@att_c) @ h -> d_out[0:64)
    gemm_k<0><<<blocks8(W64), 256, 0, stream>>>(Mc16, hT16, nullptr, dout, nullptr);
    // g-branch step 1: g_W @ h -> transposed f16 HgB
    gemm_k<1><<<blocks8(W64), 256, 0, stream>>>(gW16, hT16, nullptr, dout, HgB);
    // g-branch step 2: att_g @ HgB -> d_out[128:192)
    gemm_k<5><<<blocks8((long)Bn * 20 * (EG / 64)), 256, 0, stream>>>(attg16, HgB, nullptr,
                                                                      dout, nullptr);
    // t-branch: attention apply (VALU) then two conv GEMMs
    k_tapply<<<Bn * 77, 256, 0, stream>>>(h, att_t, tcatT);
    gemm_k<2><<<blocks8(W64), 256, 0, stream>>>(W0cat16, tcatT, tc_b0, dout, y0catT);
    gemm_k<3><<<blocks8(W64), 256, 0, stream>>>(W1cat16, y0catT, nullptr, dout, nullptr);
    // residual GEMM + bias + branch sum + ReLU, in-place on d_out
    gemm_k<4><<<blocks8((long)Bn * 12 * (NTPB / 64)), 256, 0, stream>>>(resW16, xT16, resbias,
                                                                        dout, nullptr);
}
